// VectorQuantizer_790273982748
// MI455X (gfx1250) — matrix operand order repaired and checked by reading, not compiled, not run on hardware
//
#include <hip/hip_runtime.h>

typedef __attribute__((ext_vector_type(16))) _Float16 v16h;
typedef __attribute__((ext_vector_type(8)))  float    v8f;

#define EMBED_DIM 64
#define CODES 512
#define BSTRIDE 72                 // halfs per code row in LDS (padded: 144B, 16B-aligned, bank-spread)
#define ROWS_PER_WAVE 16
#define WAVES_PER_BLOCK 8
#define ROWS_PER_BLOCK (ROWS_PER_WAVE * WAVES_PER_BLOCK)

__global__ __launch_bounds__(256)
void vq_main(const float* __restrict__ X,
             const float* __restrict__ E,
             float* __restrict__ outQ,
             float* __restrict__ outIdx,
             float* __restrict__ partial)
{
    __shared__ __align__(16) _Float16 sB[CODES * BSTRIDE];   // f16 codebook, padded rows
    __shared__ float sEsq[CODES];
    __shared__ float sPart[WAVES_PER_BLOCK];

    const int tid = threadIdx.x;

    // ---- stage codebook (f32 -> f16) into LDS ----
    for (int p = tid; p < CODES * EMBED_DIM / 2; p += 256) {
        float2 v = ((const float2*)E)[p];
        int e2 = p * 2;
        int code = e2 >> 6, dim = e2 & 63;
        sB[code * BSTRIDE + dim]     = (_Float16)v.x;
        sB[code * BSTRIDE + dim + 1] = (_Float16)v.y;
    }
    // ---- |e|^2 per code (f32) ----
    for (int c = tid; c < CODES; c += 256) {
        const float* er = E + c * EMBED_DIM;
        float s = 0.f;
        #pragma unroll
        for (int d = 0; d < EMBED_DIM; ++d) s = fmaf(er[d], er[d], s);
        sEsq[c] = s;
    }
    __syncthreads();

    const int lane = tid & 31;
    const int wave = tid >> 5;
    const int rowBase = blockIdx.x * ROWS_PER_BLOCK + wave * ROWS_PER_WAVE;
    const int col = lane & 15;     // B/C/D column (N) and A row (M) for this lane
    const int hh  = lane >> 4;     // which 16-lane half
    const int kb  = hh * 8;        // A-layout K chunk base

    // ---- load 16x64 A tile as f16 in WMMA A layout ----
    // lane<16: row=col, K elems {0..7,16..23}(+ktile*32); lane>=16: K elems {8..15,24..31}
    const float* xr = X + (size_t)(rowBase + col) * EMBED_DIM;
    v16h a0, a1;
    #pragma unroll
    for (int i = 0; i < 8; ++i) {
        a0[i]     = (_Float16)xr[kb + i];
        a0[i + 8] = (_Float16)xr[kb + 16 + i];
        a1[i]     = (_Float16)xr[kb + 32 + i];
        a1[i + 8] = (_Float16)xr[kb + 48 + i];
    }

    float bestd[8];
    int   besti[8];
    #pragma unroll
    for (int v = 0; v < 8; ++v) { bestd[v] = 3.4e38f; besti[v] = 0; }

    // ---- distance GEMM: 32 tiles of 16 codes, 2 WMMA k-steps each ----
    #pragma unroll 2
    for (int nb = 0; nb < CODES / 16; ++nb) {
        const int code = nb * 16 + col;
        // B layout 32x16: lanes 0-15 hold K {0..7,16..23} of col, lanes 16-31 K {8..15,24..31}
        const _Float16* bp0 = &sB[code * BSTRIDE + hh * 8];
        const _Float16* bp1 = bp0 + 32;
        v16h b0, b1;
        #pragma unroll
        for (int i = 0; i < 16; ++i) { b0[i] = bp0[(i & 7) + ((i & 8) << 1)]; b1[i] = bp1[(i & 7) + ((i & 8) << 1)]; }

        v8f c = {};
        c = __builtin_amdgcn_wmma_f32_16x16x32_f16(false, a0, false, b0, (short)0, c, false, false);
        c = __builtin_amdgcn_wmma_f32_16x16x32_f16(false, a1, false, b1, (short)0, c, false, false);

        const float esq = sEsq[code];
        #pragma unroll
        for (int v = 0; v < 8; ++v) {
            float dval = fmaf(-2.f, c[v], esq);     // |e|^2 - 2 x.e  (|x|^2 constant per row)
            if (dval < bestd[v]) { bestd[v] = dval; besti[v] = code; }
        }
    }

    // ---- cross-lane argmin within each 16-lane half (row M = v + 8*hh) ----
    #pragma unroll
    for (int v = 0; v < 8; ++v) {
        float bd = bestd[v]; int bi = besti[v];
        #pragma unroll
        for (int m = 1; m < 16; m <<= 1) {
            float od = __shfl_xor(bd, m, 32);
            int   oi = __shfl_xor(bi, m, 32);
            if (od < bd || (od == bd && oi < bi)) { bd = od; bi = oi; }
        }
        bestd[v] = bd; besti[v] = bi;
    }
    if (col == 0) {
        #pragma unroll
        for (int v = 0; v < 8; ++v)
            outIdx[rowBase + hh * 8 + v] = (float)besti[v];
    }

    // ---- gather quantized rows (f32), straight-through output, error accum ----
    float err = 0.f;
    #pragma unroll
    for (int r = 0; r < 16; ++r) {
        int qidx = __shfl(besti[r & 7], (r & 8) ? 16 : 0, 32);
        const float2 e = ((const float2*)(E + (size_t)qidx * EMBED_DIM))[lane];
        const float2 x = ((const float2*)(X + (size_t)(rowBase + r) * EMBED_DIM))[lane];
        float d0 = e.x - x.x, d1 = e.y - x.y;
        float2 q; q.x = x.x + d0; q.y = x.y + d1;    // matches x + stopgrad(q - x)
        ((float2*)(outQ + (size_t)(rowBase + r) * EMBED_DIM))[lane] = q;
        err = fmaf(d0, d0, err);
        err = fmaf(d1, d1, err);
    }
    #pragma unroll
    for (int m = 16; m >= 1; m >>= 1) err += __shfl_xor(err, m, 32);
    if (lane == 0) sPart[wave] = err;
    __syncthreads();
    if (tid == 0) {
        float s = 0.f;
        #pragma unroll
        for (int w = 0; w < WAVES_PER_BLOCK; ++w) s += sPart[w];
        partial[blockIdx.x] = s;
    }
}

__global__ void vq_finalize(const float* __restrict__ partial, int nPart,
                            float* __restrict__ loss, float scale)
{
    if (threadIdx.x == 0 && blockIdx.x == 0) {
        float s = 0.f;
        for (int i = 0; i < nPart; ++i) s += partial[i];   // fixed order: deterministic
        loss[0] = s * scale;                               // (1 + 0.25) / numel
    }
}

extern "C" void kernel_launch(void* const* d_in, const int* in_sizes, int n_in,
                              void* d_out, int out_size, void* d_ws, size_t ws_size,
                              hipStream_t stream)
{
    const float* X = (const float*)d_in[0];   // [16,64,64,64] f32
    const float* E = (const float*)d_in[1];   // [512,64] f32

    const int nElem  = in_sizes[0];              // 4,194,304
    const int nRows  = nElem / EMBED_DIM;        // 65,536
    const int nBlocks = nRows / ROWS_PER_BLOCK;  // 512

    float* out    = (float*)d_out;
    float* outQ   = out + 1;                     // quantized_st, nElem floats
    float* outIdx = out + 1 + nElem;             // indices (as float), nRows values
    float* partial = (float*)d_ws;               // nBlocks partial sums

    vq_main<<<nBlocks, 256, 0, stream>>>(X, E, outQ, outIdx, partial);

    const float scale = 1.25f / (float)nElem;
    vq_finalize<<<1, 32, 0, stream>>>(partial, nBlocks, out, scale);
}
